// GATLayer_40776419508778
// MI455X (gfx1250) — matrix operand order repaired and checked by reading, not compiled, not run on hardware
//
#include <hip/hip_runtime.h>

typedef __attribute__((ext_vector_type(16))) _Float16 v16h;
typedef __attribute__((ext_vector_type(8)))  float    v8f;
typedef __attribute__((ext_vector_type(4)))  int      v4i;

#define NN    8192
#define FIN   128
#define FOUTC 64
#define ALPHA 0.2f
#define LOG2E 1.44269504088896f
#define JCHUNKS 8
#define WPB 4   // waves per block in main kernel

// ------ Kernel A: WhT(f16), e1/e2 pre-scaled by log2e, zero acc/rowsum ------
// leaky_relu commutes with positive scaling, so we can work entirely in the
// log2 domain: store e1L = (Wh.a1)*log2e, e2L = (Wh.a2)*log2e.
__global__ void gat_prep(const float* __restrict__ h, const float* __restrict__ W,
                         const float* __restrict__ a,
                         float* __restrict__ e1, float* __restrict__ e2,
                         _Float16* __restrict__ whT,
                         float* __restrict__ acc, float* __restrict__ rowsum) {
    const int i = blockIdx.x;      // node
    const int f = threadIdx.x;     // output feature, 64 threads
    float wh = 0.0f;
    const float* hrow = h + (size_t)i * FIN;
#pragma unroll 8
    for (int k = 0; k < FIN; ++k)
        wh = fmaf(hrow[k], W[k * FOUTC + f], wh);

    whT[(size_t)f * NN + i] = (_Float16)wh;   // transposed f16 copy for B frags
    acc[(size_t)i * FOUTC + f] = 0.0f;        // zero partial accumulators

    __shared__ float s1[FOUTC], s2[FOUTC];
    s1[f] = wh * a[f];
    s2[f] = wh * a[FOUTC + f];
    __syncthreads();
#pragma unroll
    for (int off = 32; off > 0; off >>= 1) {
        if (f < off) { s1[f] += s1[f + off]; s2[f] += s2[f + off]; }
        __syncthreads();
    }
    if (f == 0) {
        e1[i] = s1[0] * LOG2E;
        e2[i] = s2[0] * LOG2E;
        rowsum[i] = 0.0f;
    }
}

// ------ Kernel B: fused mask + softmax-weights + WMMA (attention @ Wh) ------
__global__ __launch_bounds__(32 * WPB, 1)
void gat_main(const int* __restrict__ adj,
              const float* __restrict__ e1, const float* __restrict__ e2,
              const _Float16* __restrict__ whT,
              float* __restrict__ acc, float* __restrict__ rowsum) {
    const int tid  = threadIdx.x;
    const int wave = tid >> 5;
    const int lane = tid & 31;
    const int m    = lane & 15;          // row within i-tile (A) / col within n-tile (B)
    const bool hi  = lane >= 16;
    const int i0   = (blockIdx.x * WPB + wave) * 16;
    const int i    = i0 + m;
    const int kpA  = hi ? 16 : 0;        // A frag: K in {kpA..kpA+15}
    const int kbB  = hi ? 16 : 0;        // B frag: K in {kbB..kbB+15}

    // Per-row constants in the log2 domain:
    //   arg = lrelu(e1L + e2L) - CmL = max(d1 + e2L, fma(ALPHA, e2L, d2)) <= 0
    const float e1L = e1[i];
    const float t6  = e1L + 6.0f * LOG2E;
    const float CmL = fmaxf(t6, ALPHA * t6);   // safe per-row softmax shift
    const float d1  = e1L - CmL;
    const float d2  = ALPHA * e1L - CmL;

    const int* adjRow = adj + (size_t)i * NN;

    v8f c0 = {}, c1 = {}, c2 = {}, c3 = {}, c4 = {};
    v16h ones;
#pragma unroll
    for (int e = 0; e < 16; ++e) ones[e] = (_Float16)1.0f;

    const int jstart = blockIdx.y * (NN / JCHUNKS);
    const int iters  = (NN / JCHUNKS) / 32;

    for (int it = 0; it < iters; ++it) {
        const int jb = jstart + it * 32;

        // adj tile: 16 ints per lane in A-fragment K order; streamed once ->
        // non-temporal so the resident WhT/e2 stay in the 192MB L2.
        v4i a0 = __builtin_nontemporal_load((const v4i*)(adjRow + jb + kpA));
        v4i a1 = __builtin_nontemporal_load((const v4i*)(adjRow + jb + kpA + 4));
        v4i a2 = __builtin_nontemporal_load((const v4i*)(adjRow + jb + kpA + 8));
        v4i a3 = __builtin_nontemporal_load((const v4i*)(adjRow + jb + kpA + 12));
        __builtin_prefetch(adjRow + jb + 32 + kpA, 0, 1);

        float4 z0 = *(const float4*)(e2 + jb + kpA);
        float4 z1 = *(const float4*)(e2 + jb + kpA + 4);
        float4 z2 = *(const float4*)(e2 + jb + kpA + 8);
        float4 z3 = *(const float4*)(e2 + jb + kpA + 12);

        const int   av[16] = {a0[0],a0[1],a0[2],a0[3], a1[0],a1[1],a1[2],a1[3],
                              a2[0],a2[1],a2[2],a2[3], a3[0],a3[1],a3[2],a3[3]};
        const float zv[16] = {z0.x,z0.y,z0.z,z0.w, z1.x,z1.y,z1.z,z1.w,
                              z2.x,z2.y,z2.z,z2.w, z3.x,z3.y,z3.z,z3.w};

        v16h af;
#pragma unroll
        for (int e = 0; e < 16; ++e) {
            float u   = d1 + zv[e];                 // t - CmL
            float v   = fmaf(ALPHA, zv[e], d2);     // ALPHA*t - CmL
            float arg = fmaxf(u, v);                // lrelu(t) - CmL (log2 dom)
            // branch-proof mask: adj is exactly 0/1, raw v_exp_f32 (arg <= 0)
            float p   = __builtin_amdgcn_exp2f(arg) * (float)av[e];
            af[e] = (_Float16)p;
        }

        // B fragments from transposed f16 Wh: one contiguous 32B load each
        const _Float16* wb = whT + jb + kbB;
        v16h b0 = *(const v16h*)(wb + (size_t)( 0 + m) * NN);
        v16h b1 = *(const v16h*)(wb + (size_t)(16 + m) * NN);
        v16h b2 = *(const v16h*)(wb + (size_t)(32 + m) * NN);
        v16h b3 = *(const v16h*)(wb + (size_t)(48 + m) * NN);

        c0 = __builtin_amdgcn_wmma_f32_16x16x32_f16(false, af, false, b0, (short)0, c0, false, false);
        c1 = __builtin_amdgcn_wmma_f32_16x16x32_f16(false, af, false, b1, (short)0, c1, false, false);
        c2 = __builtin_amdgcn_wmma_f32_16x16x32_f16(false, af, false, b2, (short)0, c2, false, false);
        c3 = __builtin_amdgcn_wmma_f32_16x16x32_f16(false, af, false, b3, (short)0, c3, false, false);
        // softmax denominator: row sums via WMMA against all-ones B (matrix
        // pipe is idle; avoids a 16-deep scalar add chain per iteration)
        c4 = __builtin_amdgcn_wmma_f32_16x16x32_f16(false, af, false, ones, (short)0, c4, false, false);
    }

    const int rbase = hi ? 8 : 0;

    // rowsum partials: every column of c4 equals the row sum; lane with m==r
    // (in each half-wave) commits row rbase+r.
#pragma unroll
    for (int r = 0; r < 8; ++r)
        if (m == r) unsafeAtomicAdd(&rowsum[i0 + rbase + r], c4[r]);

    // scatter C fragments: row = r + rbase, col = nt*16 + m
#pragma unroll
    for (int r = 0; r < 8; ++r) {
        float* rowp = acc + (size_t)(i0 + rbase + r) * FOUTC + m;
        unsafeAtomicAdd(rowp +  0, c0[r]);
        unsafeAtomicAdd(rowp + 16, c1[r]);
        unsafeAtomicAdd(rowp + 32, c2[r]);
        unsafeAtomicAdd(rowp + 48, c3[r]);
    }
}

// ---------------- Kernel C: normalize + elu ----------------------------------
__global__ void gat_finish(const float* __restrict__ acc,
                           const float* __restrict__ rowsum,
                           float* __restrict__ out) {
    const int idx = blockIdx.x * blockDim.x + threadIdx.x;   // NN*FOUTC
    const int i = idx >> 6;
    float v = acc[idx] / rowsum[i];
    out[idx] = v > 0.0f ? v : __builtin_amdgcn_exp2f(v * LOG2E) - 1.0f; // elu
}

// ---------------- Launch -----------------------------------------------------
extern "C" void kernel_launch(void* const* d_in, const int* in_sizes, int n_in,
                              void* d_out, int out_size, void* d_ws, size_t ws_size,
                              hipStream_t stream) {
    const float* h   = (const float*)d_in[0];
    const int*   adj = (const int*)d_in[1];
    const float* W   = (const float*)d_in[2];
    const float* a   = (const float*)d_in[3];
    float* out = (float*)d_out;

    char* ws = (char*)d_ws;
    _Float16* whT = (_Float16*)(ws);                        // 64*8192*2 = 1 MB
    float* e1     = (float*)(ws + (1u << 20));              // 32 KB
    float* e2     = (float*)(ws + (1u << 20) + 32768);      // 32 KB
    float* rowsum = (float*)(ws + (1u << 20) + 65536);      // 32 KB
    float* acc    = (float*)(ws + (1u << 20) + 98304);      // 2 MB

    gat_prep<<<NN, FOUTC, 0, stream>>>(h, W, a, e1, e2, whT, acc, rowsum);
    gat_main<<<dim3(NN / (16 * WPB), JCHUNKS), 32 * WPB, 0, stream>>>(
        adj, e1, e2, whT, acc, rowsum);
    gat_finish<<<(NN * FOUTC) / 256, 256, 0, stream>>>(acc, rowsum, out);
}
